// Net_59949153518048
// MI455X (gfx1250) — matrix operand order repaired and checked by reading, not compiled, not run on hardware
//
#include <hip/hip_runtime.h>
#include <hip/hip_bf16.h>
#include <math.h>

// Problem constants (match reference)
#define N_NODES 16384
#define IN_DIM  13
#define HID     64
#define KNN     40
#define OUT_W   (HID + IN_DIM)   // 77

typedef _Float16 half_t;
typedef __attribute__((ext_vector_type(16))) _Float16 v16h;
typedef __attribute__((ext_vector_type(8)))  _Float16 v8h;
typedef __attribute__((ext_vector_type(8)))  float    v8f;

union V16H { v16h v; v8h h8[2]; _Float16 e[16]; };

__device__ __forceinline__ float silu_f(float x) {
    return x / (1.0f + __expf(-x));
}

// ---- CDNA5 async global->LDS copy (guarded; fallback = sync ds staging) ----
#if defined(__has_builtin)
#if __has_builtin(__builtin_amdgcn_global_load_async_to_lds_b128)
#define HAVE_ASYNC_LDS 1
#else
#define HAVE_ASYNC_LDS 0
#endif
#else
#define HAVE_ASYNC_LDS 0
#endif

// Builtin signature (from hipcc diagnostic): (int4 AS1*, int4 AS3*, Imm, Imm)
typedef int v4i_t __attribute__((vector_size(16)));
typedef __attribute__((address_space(1))) v4i_t* gptr4_t;
typedef __attribute__((address_space(3))) v4i_t* sptr4_t;

__device__ __forceinline__ void copy_b128_to_lds(const half_t* g, half_t* l) {
#if HAVE_ASYNC_LDS
    // Generic LDS pointers carry the LDS byte offset in the low 32 bits
    // (ISA 10.2 aperture rules), so the truncating inttoptr yields AS3.
    __builtin_amdgcn_global_load_async_to_lds_b128(
        (gptr4_t)(uintptr_t)g, (sptr4_t)(uintptr_t)l, 0, 0);
#else
    *(v8h*)l = *(const v8h*)g;   // sync fallback: global_load_b128 + ds_store
#endif
}

__device__ __forceinline__ void wait_async_copies() {
#if HAVE_ASYNC_LDS
#if __has_builtin(__builtin_amdgcn_s_wait_asynccnt)
    __builtin_amdgcn_s_wait_asynccnt(0);
#else
    asm volatile("s_wait_asynccnt 0x0" ::: "memory");
#endif
#endif
}

// ---------------------------------------------------------------------------
// Kernel 1: per-node encoder MLP  (13 -> 32 -> 64 -> 64), SiLU between.
// Also writes h as f16 (for WMMA) and sq = |h|^2.
// ---------------------------------------------------------------------------
__global__ void k_encode(const float* __restrict__ x,
                         const float* __restrict__ W1, const float* __restrict__ b1,
                         const float* __restrict__ W2, const float* __restrict__ b2,
                         const float* __restrict__ W3, const float* __restrict__ b3,
                         float* __restrict__ h32, half_t* __restrict__ h16,
                         float* __restrict__ sq) {
    __shared__ float sW1[32 * 13], sb1[32];
    __shared__ float sW2[64 * 32], sb2[64];
    __shared__ float sW3[64 * 64], sb3[64];
    const int tid = threadIdx.x;
    for (int t = tid; t < 32 * 13; t += 256) sW1[t] = W1[t];
    for (int t = tid; t < 32;      t += 256) sb1[t] = b1[t];
    for (int t = tid; t < 64 * 32; t += 256) sW2[t] = W2[t];
    for (int t = tid; t < 64;      t += 256) sb2[t] = b2[t];
    for (int t = tid; t < 64 * 64; t += 256) sW3[t] = W3[t];
    for (int t = tid; t < 64;      t += 256) sb3[t] = b3[t];
    __syncthreads();

    const int i = blockIdx.x * 256 + tid;
    float xr[IN_DIM];
#pragma unroll
    for (int k = 0; k < IN_DIM; ++k) xr[k] = x[(size_t)i * IN_DIM + k];

    float t1[32];
#pragma unroll 4
    for (int o = 0; o < 32; ++o) {
        float acc = sb1[o];
#pragma unroll
        for (int k = 0; k < IN_DIM; ++k) acc += sW1[o * IN_DIM + k] * xr[k];
        t1[o] = silu_f(acc);
    }
    float t2[HID];
#pragma unroll 4
    for (int o = 0; o < HID; ++o) {
        float acc = sb2[o];
#pragma unroll
        for (int k = 0; k < 32; ++k) acc += sW2[o * 32 + k] * t1[k];
        t2[o] = silu_f(acc);
    }
    float s = 0.0f;
#pragma unroll 4
    for (int o = 0; o < HID; ++o) {
        float acc = sb3[o];
#pragma unroll
        for (int k = 0; k < HID; ++k) acc += sW3[o * HID + k] * t2[k];
        h32[(size_t)i * HID + o] = acc;
        h16[(size_t)i * HID + o] = (half_t)acc;
        s += acc * acc;
    }
    sq[i] = s;
}

// ---------------------------------------------------------------------------
// Kernel 1b: convert edge-MLP weight We [64 x 128] to f16 for WMMA B-frags.
// ---------------------------------------------------------------------------
__global__ void k_cvtWe(const float* __restrict__ We, half_t* __restrict__ We16) {
    int t = blockIdx.x * 256 + threadIdx.x;
    if (t < HID * 2 * HID) We16[t] = (half_t)We[t];
}

// ---------------------------------------------------------------------------
// Kernel 2: fused kNN.  Each wave owns 16 rows; the block shares a 16-node
// column tile staged into LDS via double-buffered async global->LDS copies
// (ASYNCcnt).  Distance tile via 2 chained v_wmma_f32_16x16x32_f16 (K=64),
// then per-row top-40 maintained in LDS (2 lanes per row).  h16 (2 MiB) is
// L2-resident, so after the 4x dedup this loop is compute-bound.
// ---------------------------------------------------------------------------
__global__ void k_knn(const half_t* __restrict__ h16, const float* __restrict__ sq,
                      int* __restrict__ nbr) {
    __shared__ float tile[4][16][16];                       //  4 KB (wave-private slices)
    __shared__ float lV[4][32][KNN];                        // 20 KB
    __shared__ int   lI[4][32][KNN];                        // 20 KB
    __shared__ __align__(16) half_t colbuf[2][16][HID];     //  4 KB (block-shared, 2 buffers)

    const int tid  = threadIdx.x;
    const int lane = tid & 31;
    const int w    = tid >> 5;
    const int g    = lane >> 4;
    const int l15  = lane & 15;
    const int r0   = blockIdx.x * 64 + w * 16;

    // Staging assignment: 16 rows x 8 x b128 = 128 transfers = 1 per thread.
    const int srow = tid >> 3;          // 0..15
    const int spart = (tid & 7) * 8;    // half offset within row, 16B granular

    // Resident A fragments for this wave's 16 rows (ISA 16-bit A layout:
    // lane group g holds K in {g*8..g*8+7} and {16+g*8..+7} per K32 chunk).
    V16H A0, A1;
    {
        const half_t* base = h16 + (size_t)(r0 + l15) * HID;
        const v8h* p0 = (const v8h*)(base + 0 * 32 + g * 8);
        A0.h8[0] = p0[0]; A0.h8[1] = p0[2];
        const v8h* p1 = (const v8h*)(base + 1 * 32 + g * 8);
        A1.h8[0] = p1[0]; A1.h8[1] = p1[2];
    }
    float sqr[8];
#pragma unroll
    for (int v = 0; v < 8; ++v) sqr[v] = sq[r0 + v + 8 * g];

    for (int t = 0; t < KNN; ++t) lV[w][lane][t] = 3.0e38f;
    float curMax = 3.0e38f;
    int   curPos = 0;
    const int rr = lane >> 1;     // row this lane scans
    const int pp = lane & 1;      // column parity

    // Prologue: stage column tile 0 into buffer 0.
    copy_b128_to_lds(h16 + (size_t)srow * HID + spart, &colbuf[0][srow][spart]);

    for (int j0 = 0; j0 < N_NODES; j0 += 16) {
        const int cur = (j0 >> 4) & 1;
        wait_async_copies();
        __syncthreads();            // staged tile visible to all waves

        // Kick off the next tile into the other buffer (overlaps compute).
        if (j0 + 16 < N_NODES)
            copy_b128_to_lds(h16 + (size_t)(j0 + 16 + srow) * HID + spart,
                             &colbuf[cur ^ 1][srow][spart]);

        // B fragments from LDS (B layout: lane group g holds K in
        // {g*8..g*8+7} and {16+g*8..+7} per K32 chunk, matching A).
        const half_t* cb = &colbuf[cur][l15][0];
        V16H B0, B1;
        const v8h* q0 = (const v8h*)(cb + g * 8);
        B0.h8[0] = q0[0]; B0.h8[1] = q0[2];
        const v8h* q1 = (const v8h*)(cb + 32 + g * 8);
        B1.h8[0] = q1[0]; B1.h8[1] = q1[2];

        v8f c = {0.f, 0.f, 0.f, 0.f, 0.f, 0.f, 0.f, 0.f};
        c = __builtin_amdgcn_wmma_f32_16x16x32_f16(false, A0.v, false, B0.v,
                                                   (short)0, c, false, false);
        c = __builtin_amdgcn_wmma_f32_16x16x32_f16(false, A1.v, false, B1.v,
                                                   (short)0, c, false, false);

        const float sqc = sq[j0 + l15];
#pragma unroll
        for (int v = 0; v < 8; ++v)
            tile[w][v + 8 * g][l15] = sqr[v] + sqc - 2.0f * c[v];
        // d2 tile hand-off is wave-internal: LDS ops from one wave are
        // in-order (ISA 7.1); just fence the scheduler.
        __builtin_amdgcn_wave_barrier();

        // Top-K scan: each lane handles 8 of the 16 columns of its row.
#pragma unroll
        for (int t = 0; t < 8; ++t) {
            const int cc = pp + 2 * t;
            const float d = tile[w][rr][cc];
            if (d < curMax) {
                lV[w][lane][curPos] = d;
                lI[w][lane][curPos] = j0 + cc;
                float m = -3.0e38f; int mp = 0;
                for (int s = 0; s < KNN; ++s) {
                    const float vv = lV[w][lane][s];
                    if (vv > m) { m = vv; mp = s; }
                }
                curMax = m; curPos = mp;
            }
        }
    }
    __syncthreads();

    // Merge the two 40-candidate lists per row -> final 40 (order-free:
    // downstream is a mean over K, so set equality is what matters).
    if (lane < 16) {
        const int row = r0 + lane;
        for (int s = 0; s < KNN; ++s) {
            float m = 3.3e38f; int mt = 0;
            for (int t = 0; t < 2 * KNN; ++t) {
                const float vv = (t < KNN) ? lV[w][2 * lane][t]
                                           : lV[w][2 * lane + 1][t - KNN];
                if (vv < m) { m = vv; mt = t; }
            }
            int j;
            if (mt < KNN) { j = lI[w][2 * lane][mt];          lV[w][2 * lane][mt] = 3.4e38f; }
            else          { j = lI[w][2 * lane + 1][mt - KNN]; lV[w][2 * lane + 1][mt - KNN] = 3.4e38f; }
            nbr[(size_t)row * KNN + s] = j;
        }
    }
}

// ---------------------------------------------------------------------------
// Kernel 3: EdgeConv (msg = [x_i, x_j-x_i] @ We^T + be, SiLU, mean over K)
// + FFN + concat with x_pfc.  One node per wave; We^T held as 16 resident
// WMMA B-fragments; msg staged in LDS f16; 48 WMMAs per node.
// ---------------------------------------------------------------------------
__global__ void k_edge(const float* __restrict__ h32, const half_t* __restrict__ We16,
                       const float* __restrict__ be,
                       const float* __restrict__ Wf1, const float* __restrict__ bf1,
                       const float* __restrict__ Wf2, const float* __restrict__ bf2,
                       const int* __restrict__ nbr, const float* __restrict__ x,
                       float* __restrict__ out) {
    __shared__ __align__(16) half_t msg[4][48][128];   // 48 KB
    __shared__ float feats[4][64];                     //  1 KB
    __shared__ float g1[4][128];                       //  2 KB

    const int lane = threadIdx.x & 31;
    const int w    = threadIdx.x >> 5;
    const int g    = lane >> 4;
    const int l15  = lane & 15;
    const int i    = blockIdx.x * 4 + w;

    // Resident B fragments of We^T: B[k, o] = We[o, k] -> per-lane 16
    // halfs from We row o, k in {g*8..g*8+7} and {16+g*8..+7} per K32 chunk.  4 col-tiles x 4 K32-chunks.
    V16H Bf[4][4];
    float beC[4];
#pragma unroll
    for (int ot = 0; ot < 4; ++ot) {
        const int o = ot * 16 + l15;
        beC[ot] = be[o];
#pragma unroll
        for (int kt = 0; kt < 4; ++kt) {
            const v8h* p = (const v8h*)(We16 + o * 128 + kt * 32 + g * 8);
            Bf[ot][kt].h8[0] = p[0];
            Bf[ot][kt].h8[1] = p[2];
        }
    }

    // Build msg rows in LDS (f16): [h_i | h_j - h_i], rows 40..47 zero pad.
    for (int e = 0; e < KNN; ++e) {
        const int j = nbr[(size_t)i * KNN + e];
#pragma unroll
        for (int hlf = 0; hlf < 2; ++hlf) {
            const int c0 = lane + 32 * hlf;
            const float hi = h32[(size_t)i * HID + c0];
            const float hj = h32[(size_t)j * HID + c0];
            msg[w][e][c0]      = (half_t)hi;
            msg[w][e][64 + c0] = (half_t)(hj - hi);
        }
    }
    for (int e = KNN; e < 48; ++e)
        for (int c0 = lane; c0 < 128; c0 += 32)
            msg[w][e][c0] = (half_t)0.f;
    __syncthreads();

    // GEMM: msg[48x128] x We^T[128x64] via WMMA, SiLU, masked mean over rows.
    float colacc[4] = {0.f, 0.f, 0.f, 0.f};
    for (int rt = 0; rt < 3; ++rt) {
        V16H Af[4];
#pragma unroll
        for (int kt = 0; kt < 4; ++kt) {
            const v8h* p = (const v8h*)(&msg[w][rt * 16 + l15][kt * 32] + g * 8);
            Af[kt].h8[0] = p[0];
            Af[kt].h8[1] = p[2];
        }
#pragma unroll
        for (int ot = 0; ot < 4; ++ot) {
            v8f c = {0.f, 0.f, 0.f, 0.f, 0.f, 0.f, 0.f, 0.f};
#pragma unroll
            for (int kt = 0; kt < 4; ++kt)
                c = __builtin_amdgcn_wmma_f32_16x16x32_f16(
                        false, Af[kt].v, false, Bf[ot][kt].v,
                        (short)0, c, false, false);
#pragma unroll
            for (int v = 0; v < 8; ++v) {
                const int rA = rt * 16 + v + 8 * g;   // edge index
                const float m = silu_f(c[v] + beC[ot]);
                if (rA < KNN) colacc[ot] += m;
            }
        }
    }
#pragma unroll
    for (int ot = 0; ot < 4; ++ot) {
        colacc[ot] += __shfl_xor(colacc[ot], 16, 32);  // combine row halves
        if (g == 0) feats[w][ot * 16 + l15] = colacc[ot] * (1.0f / KNN);
    }
    __syncthreads();

    // FFN layer 1: 64 -> 128, SiLU (4 outputs per lane).
#pragma unroll
    for (int t = 0; t < 4; ++t) {
        const int o = lane * 4 + t;
        float acc = bf1[o];
        for (int k = 0; k < HID; ++k) acc += Wf1[o * HID + k] * feats[w][k];
        g1[w][o] = silu_f(acc);
    }
    __syncthreads();

    // FFN layer 2: 128 -> 64 (2 outputs per lane), then concat x_pfc.
#pragma unroll
    for (int t = 0; t < 2; ++t) {
        const int o = lane * 2 + t;
        float acc = bf2[o];
        for (int k = 0; k < 128; ++k) acc += Wf2[o * 128 + k] * g1[w][k];
        out[(size_t)i * OUT_W + o] = acc;
    }
    if (lane < IN_DIM)
        out[(size_t)i * OUT_W + HID + lane] = x[(size_t)i * IN_DIM + lane];
}

// ---------------------------------------------------------------------------
// Host launcher
// ---------------------------------------------------------------------------
extern "C" void kernel_launch(void* const* d_in, const int* in_sizes, int n_in,
                              void* d_out, int out_size, void* d_ws, size_t ws_size,
                              hipStream_t stream) {
    (void)in_sizes; (void)n_in; (void)out_size; (void)ws_size;
    const float* x   = (const float*)d_in[0];
    const float* W1  = (const float*)d_in[1];
    const float* b1  = (const float*)d_in[2];
    const float* W2  = (const float*)d_in[3];
    const float* b2  = (const float*)d_in[4];
    const float* W3  = (const float*)d_in[5];
    const float* b3  = (const float*)d_in[6];
    const float* We  = (const float*)d_in[7];
    const float* be  = (const float*)d_in[8];
    const float* Wf1 = (const float*)d_in[9];
    const float* bf1 = (const float*)d_in[10];
    const float* Wf2 = (const float*)d_in[11];
    const float* bf2 = (const float*)d_in[12];

    char* ws = (char*)d_ws;
    float*  h32  = (float*) (ws + 0);                        // 4 MiB
    half_t* h16  = (half_t*)(ws + 4u * 1024 * 1024);         // 2 MiB
    float*  sq   = (float*) (ws + 6u * 1024 * 1024);         // 64 KiB
    int*    nbr  = (int*)   (ws + 6u * 1024 * 1024 + 256u * 1024); // 2.5 MiB
    half_t* We16 = (half_t*)(ws + 9u * 1024 * 1024);         // 16 KiB

    k_encode<<<N_NODES / 256, 256, 0, stream>>>(x, W1, b1, W2, b2, W3, b3,
                                                h32, h16, sq);
    k_cvtWe<<<(HID * 2 * HID + 255) / 256, 256, 0, stream>>>(We, We16);
    k_knn<<<N_NODES / 64, 128, 0, stream>>>(h16, sq, nbr);
    k_edge<<<N_NODES / 4, 128, 0, stream>>>(h32, We16, be, Wf1, bf1, Wf2, bf2,
                                            nbr, x, (float*)d_out);
}
